// SparseEncoderResUNet_16758962389330
// MI455X (gfx1250) — matrix operand order repaired and checked by reading, not compiled, not run on hardware
//
#include <hip/hip_runtime.h>
#include <hip/hip_bf16.h>
#include <cstdint>
#include <cstddef>

// ---------------------------------------------------------------------------
// CDNA5 (gfx1250) sparse 3D conv encoder.
// Implicit-GEMM convs on v_wmma_f32_16x16x32_f16 (wave32, f32 accum).
// Activations: NDHWC f16 in workspace. Weights repacked to [tap][OC][ICpad]
// f16, then async-staged per tap into LDS (global_load_async_to_lds_b128,
// double buffered, s_wait_asynccnt + workgroup barrier) and shared by the
// 8 waves of each workgroup. Each wave computes a 16-voxel x 32-outch tile.
// ---------------------------------------------------------------------------

typedef _Float16 v16h __attribute__((ext_vector_type(16)));
typedef _Float16 v8h  __attribute__((ext_vector_type(8)));
typedef float    v8f  __attribute__((ext_vector_type(8)));

#define LRELU_SLOPE 0.2f

__device__ __forceinline__ float leakyf(float v) {
    return v >= 0.0f ? v : LRELU_SLOPE * v;
}

__device__ __forceinline__ void async_stage_b128(const void* gsrc, void* lds_dst) {
    // VDST = LDS byte address (low 32 bits of the generic shared pointer),
    // VADDR = 64-bit global address, saddr = off  (GV mode, 08_async_tensor §4.4)
    uint32_t loff = (uint32_t)(uintptr_t)lds_dst;
    uint64_t g    = (uint64_t)(uintptr_t)gsrc;
    asm volatile("global_load_async_to_lds_b128 %0, %1, off"
                 :: "v"(loff), "v"(g) : "memory");
}

__device__ __forceinline__ void wait_async0() {
    asm volatile("s_wait_asynccnt 0" ::: "memory");
}

// ---- pack x: NCDHW fp32 (3 ch, pre-masked) -> NDHWC f16 padded to 32 ch ----
__global__ void pack_x_kernel(const float* __restrict__ x,
                              _Float16* __restrict__ out,
                              int B, int V) {
    int idx = blockIdx.x * blockDim.x + threadIdx.x;
    if (idx >= B * V) return;
    int b  = idx / V;
    int sp = idx - b * V;
    _Float16* o = out + (size_t)idx * 32;
#pragma unroll
    for (int c = 0; c < 32; ++c) {
        float v = 0.0f;
        if (c < 3) v = x[((size_t)(b * 3 + c)) * V + sp];
        o[c] = (_Float16)v;
    }
}

// ---- pack weights: OIDHW fp32 -> [tap][OC][ICP] f16 (ic zero-padded) ----
__global__ void pack_w_kernel(const float* __restrict__ w,
                              _Float16* __restrict__ wp,
                              int OC, int IC, int ICP, int taps) {
    int idx = blockIdx.x * blockDim.x + threadIdx.x;
    int total = taps * OC * ICP;
    if (idx >= total) return;
    int icp = idx % ICP;
    int oc  = (idx / ICP) % OC;
    int t   = idx / (ICP * OC);
    float v = 0.0f;
    if (icp < IC) v = w[((size_t)(oc * IC + icp)) * taps + t];
    wp[idx] = (_Float16)v;
}

// ---- mask maxpool, k=3 s=2 p=1 ----
__global__ void mask_down_kernel(const float* __restrict__ mi,
                                 float* __restrict__ mo,
                                 int B, int Din) {
    int Do = Din >> 1;
    int Vo = Do * Do * Do;
    int idx = blockIdx.x * blockDim.x + threadIdx.x;
    if (idx >= B * Vo) return;
    int b = idx / Vo;
    int r = idx - b * Vo;
    int ox = r % Do;
    int oy = (r / Do) % Do;
    int oz = r / (Do * Do);
    float m = 0.0f;
    for (int kd = 0; kd < 3; ++kd) {
        int iz = oz * 2 + kd - 1;
        if ((unsigned)iz >= (unsigned)Din) continue;
        for (int kh = 0; kh < 3; ++kh) {
            int iy = oy * 2 + kh - 1;
            if ((unsigned)iy >= (unsigned)Din) continue;
            for (int kw = 0; kw < 3; ++kw) {
                int ix = ox * 2 + kw - 1;
                if ((unsigned)ix >= (unsigned)Din) continue;
                m = fmaxf(m, mi[(size_t)b * Din * Din * Din +
                                ((size_t)iz * Din + iy) * Din + ix]);
            }
        }
    }
    mo[idx] = m;
}

// ---------------------------------------------------------------------------
// Implicit-GEMM 3x3x3 conv, pad 1. Workgroup = 256 threads (8 waves).
// Each wave: 16 voxels x 32 out-channels (2 accumulators sharing one A frag).
// Per-tap weights [32oc x ICP] are async-staged to LDS, double buffered.
//   RES == 0: plain;  RES == 1: += elementwise residual (f16 buffer);
//   RES == 2: += 1x1 conv of a 32-ch buffer (stage-1 identity path).
// Epilogue: (acc [+ 1x1]) * mask [+ residual], leaky-relu, f16 store.
// ---------------------------------------------------------------------------
template <int ICP, int OC, int STRIDE, bool MASK, int RES>
__global__ void __launch_bounds__(256)
conv3d_wmma_kernel(const _Float16* __restrict__ in,
                   const _Float16* __restrict__ wp,
                   const float* __restrict__ mask,
                   const _Float16* __restrict__ resin,
                   const _Float16* __restrict__ resw,
                   _Float16* __restrict__ out,
                   int Din) {
    const int Do = (STRIDE == 2) ? (Din >> 1) : Din;
    const int Vo = Do * Do * Do;

    const int tid  = threadIdx.x;
    const int lane = tid & 31;
    const int wave = tid >> 5;
    const int row  = lane & 15;   // A row / B column / D column
    const int kh2  = lane >> 4;   // K-half selector
    const int b      = blockIdx.z;
    const int ocbase = blockIdx.y * 32;
    const int mt     = blockIdx.x * 8 + wave;

    __shared__ __align__(16) _Float16 sW[2][32 * ICP];

    const int vmA = mt * 16 + row;              // voxel of this lane's A row
    const int ox = vmA % Do;
    const int oy = (vmA / Do) % Do;
    const int oz = vmA / (Do * Do);

    v8f acc0 = {};
    v8f acc1 = {};
    const size_t inB = (size_t)b * Din * Din * Din;

    // async-stage tap-t weight slice (32*ICP halfs, contiguous) into sW[buf]
    auto stage = [&](int buf, int t) {
        const _Float16* src = wp + ((size_t)t * OC + ocbase) * ICP;
        const int nchunk = (32 * ICP) / 8;      // 16-byte chunks
        for (int i = tid; i < nchunk; i += 256)
            async_stage_b128(src + i * 8, &sW[buf][i * 8]);
    };

    stage(0, 0);
    wait_async0();
    __syncthreads();

#pragma unroll 1
    for (int t = 0; t < 27; ++t) {
        if (t + 1 < 27) stage((t + 1) & 1, t + 1);   // overlap fetch of t+1

        const int kd = t / 9, kh = (t / 3) % 3, kw = t % 3;
        const int iz = oz * STRIDE + kd - 1;
        const int iy = oy * STRIDE + kh - 1;
        const int ix = ox * STRIDE + kw - 1;
        const bool inb = (unsigned)iz < (unsigned)Din &&
                         (unsigned)iy < (unsigned)Din &&
                         (unsigned)ix < (unsigned)Din;
        const _Float16* ap  = in + (inB + ((size_t)iz * Din + iy) * Din + ix) * ICP;
        const _Float16* bls = &sW[t & 1][0];
        __builtin_prefetch(ap + ICP, 0, 1);          // neighbor voxel (next tap)

#pragma unroll
        for (int kc = 0; kc < ICP; kc += 32) {
            v16h a;
            if (inb) {
                v8h alo = *(const v8h*)(ap + kc + kh2 * 8);
                v8h ahi = *(const v8h*)(ap + kc + 16 + kh2 * 8);
#pragma unroll
                for (int j = 0; j < 8; ++j) { a[j] = alo[j]; a[j + 8] = ahi[j]; }
            } else {
#pragma unroll
                for (int j = 0; j < 16; ++j) a[j] = (_Float16)0.0f;
            }
            // two B columns per lane (oc row and row+16) from LDS
            v8h b0lo = *(const v8h*)(bls + (size_t)row * ICP + kc + kh2 * 8);
            v8h b0hi = *(const v8h*)(bls + (size_t)row * ICP + kc + 16 + kh2 * 8);
            v8h b1lo = *(const v8h*)(bls + (size_t)(row + 16) * ICP + kc + kh2 * 8);
            v8h b1hi = *(const v8h*)(bls + (size_t)(row + 16) * ICP + kc + 16 + kh2 * 8);
            v16h b0, b1;
#pragma unroll
            for (int j = 0; j < 8; ++j) {
                b0[j] = b0lo[j]; b0[j + 8] = b0hi[j];
                b1[j] = b1lo[j]; b1[j + 8] = b1hi[j];
            }
            acc0 = __builtin_amdgcn_wmma_f32_16x16x32_f16(
                false, a, false, b0, (short)0, acc0, false, false);
            acc1 = __builtin_amdgcn_wmma_f32_16x16x32_f16(
                false, a, false, b1, (short)0, acc1, false, false);
        }

        wait_async0();       // stage(t+1) landed in LDS
        __syncthreads();     // visible to all 8 waves; buf (t&1) free to reuse
    }

    if (RES == 2) {
        // fused 1x1 identity conv (stride-1 blocks): 32-ch input, 2 WMMAs
        const _Float16* ap = resin + ((size_t)b * Vo + vmA) * 32;
        v8h alo = *(const v8h*)(ap + kh2 * 8);
        v8h ahi = *(const v8h*)(ap + 16 + kh2 * 8);
        v16h a;
#pragma unroll
        for (int j = 0; j < 8; ++j) { a[j] = alo[j]; a[j + 8] = ahi[j]; }
        const _Float16* bp0 = resw + (size_t)(ocbase + row) * 32;
        const _Float16* bp1 = resw + (size_t)(ocbase + row + 16) * 32;
        v8h b0lo = *(const v8h*)(bp0 + kh2 * 8);
        v8h b0hi = *(const v8h*)(bp0 + 16 + kh2 * 8);
        v8h b1lo = *(const v8h*)(bp1 + kh2 * 8);
        v8h b1hi = *(const v8h*)(bp1 + 16 + kh2 * 8);
        v16h b0, b1;
#pragma unroll
        for (int j = 0; j < 8; ++j) {
            b0[j] = b0lo[j]; b0[j + 8] = b0hi[j];
            b1[j] = b1lo[j]; b1[j + 8] = b1hi[j];
        }
        acc0 = __builtin_amdgcn_wmma_f32_16x16x32_f16(
            false, a, false, b0, (short)0, acc0, false, false);
        acc1 = __builtin_amdgcn_wmma_f32_16x16x32_f16(
            false, a, false, b1, (short)0, acc1, false, false);
    }

    // ---- epilogue: mask, residual, leaky, f16 store (NDHWC) ----
    const int oc0 = ocbase + row;
#pragma unroll
    for (int r = 0; r < 8; ++r) {
        const int vm = mt * 16 + r + 8 * kh2;
        float v0 = acc0[r];
        float v1 = acc1[r];
        if (MASK) {
            float m = mask[(size_t)b * Vo + vm];
            v0 *= m; v1 *= m;
        }
        if (RES == 1) {
            v0 += (float)resin[((size_t)b * Vo + vm) * OC + oc0];
            v1 += (float)resin[((size_t)b * Vo + vm) * OC + oc0 + 16];
        }
        v0 = leakyf(v0);
        v1 = leakyf(v1);
        out[((size_t)b * Vo + vm) * OC + oc0]      = (_Float16)v0;
        out[((size_t)b * Vo + vm) * OC + oc0 + 16] = (_Float16)v1;
    }
}

// ---- final global mean+max pool: [B][V][C] f16 -> [B][2C] f32 ----
__global__ void __launch_bounds__(32)
pool_kernel(const _Float16* __restrict__ in, float* __restrict__ out,
            int V, int C) {
    int c = blockIdx.x;
    int b = blockIdx.y;
    int lane = threadIdx.x;
    float s = 0.0f;
    float mx = -__builtin_inff();
    for (int v = lane; v < V; v += 32) {
        float x = (float)in[((size_t)b * V + v) * C + c];
        s += x;
        mx = fmaxf(mx, x);
    }
    for (int off = 16; off > 0; off >>= 1) {
        s += __shfl_down(s, off, 32);
        mx = fmaxf(mx, __shfl_down(mx, off, 32));
    }
    if (lane == 0) {
        out[(size_t)b * 2 * C + c]     = s / (float)V;
        out[(size_t)b * 2 * C + C + c] = mx;
    }
}

// ---------------------------------------------------------------------------
extern "C" void kernel_launch(void* const* d_in, const int* in_sizes, int n_in,
                              void* d_out, int out_size, void* d_ws, size_t ws_size,
                              hipStream_t stream) {
    (void)in_sizes; (void)n_in; (void)out_size; (void)ws_size;

    const float* x   = (const float*)d_in[0];
    const float* m1  = (const float*)d_in[1];
    const float* w1a = (const float*)d_in[2];
    const float* w1b = (const float*)d_in[3];
    const float* w1d = (const float*)d_in[4];
    const float* wd1 = (const float*)d_in[5];
    const float* w2a = (const float*)d_in[6];
    const float* w2b = (const float*)d_in[7];
    const float* wd2 = (const float*)d_in[8];
    const float* w3a = (const float*)d_in[9];
    const float* w3b = (const float*)d_in[10];
    const float* wd3 = (const float*)d_in[11];
    float* out = (float*)d_out;

    const int B = 2;
    const int V1 = 64 * 64 * 64, V2 = 32 * 32 * 32, V3 = 16 * 16 * 16, V4 = 8 * 8 * 8;

    char* ws = (char*)d_ws;
    size_t cur = 0;
    auto alloc = [&](size_t bytes) -> size_t {
        size_t o = cur;
        cur += (bytes + 255) & ~(size_t)255;
        return o;
    };

    // persistent: packed weights + mask pyramids
    size_t o_w1aP = alloc((size_t)27 * 64 * 32 * 2);
    size_t o_w1bP = alloc((size_t)27 * 64 * 64 * 2);
    size_t o_w1dP = alloc((size_t)1 * 64 * 32 * 2);
    size_t o_wd1P = alloc((size_t)27 * 128 * 64 * 2);
    size_t o_w2aP = alloc((size_t)27 * 128 * 128 * 2);
    size_t o_w2bP = alloc((size_t)27 * 128 * 128 * 2);
    size_t o_wd2P = alloc((size_t)27 * 256 * 128 * 2);
    size_t o_w3aP = alloc((size_t)27 * 256 * 256 * 2);
    size_t o_w3bP = alloc((size_t)27 * 256 * 256 * 2);
    size_t o_wd3P = alloc((size_t)27 * 128 * 256 * 2);
    size_t o_m2   = alloc((size_t)B * V2 * 4);
    size_t o_m3   = alloc((size_t)B * V3 * 4);
    // big activation regions (f16); later stages overlay dead regions
    size_t o_A0 = alloc((size_t)B * V1 * 32 * 2);   // packed x     (33.5 MB)
    size_t o_T1 = alloc((size_t)B * V1 * 64 * 2);   // stage1 tmp   (67 MB)
    size_t o_X1 = alloc((size_t)B * V1 * 64 * 2);   // stage1 out   (67 MB)
    size_t o_B2 = o_A0;                                   // 128ch @32^3
    size_t o_T2 = o_A0 + (size_t)B * V2 * 128 * 2;        // 128ch @32^3
    size_t o_X2 = o_T1;                                   // 128ch @32^3
    size_t o_B3 = o_X1;                                   // 256ch @16^3
    size_t o_T3 = o_X1 + 1 * (size_t)B * V3 * 256 * 2;
    size_t o_X3 = o_X1 + 2 * (size_t)B * V3 * 256 * 2;
    size_t o_B4 = o_X1 + 3 * (size_t)B * V3 * 256 * 2;    // 128ch @8^3

    auto H = [&](size_t o) { return (_Float16*)(ws + o); };
    auto F = [&](size_t o) { return (float*)(ws + o); };

    // ---- weight repack ----
    auto PW = [&](const float* w, size_t off, int OC, int IC, int ICP, int taps) {
        int total = taps * OC * ICP;
        pack_w_kernel<<<(total + 255) / 256, 256, 0, stream>>>(w, H(off), OC, IC, ICP, taps);
    };
    PW(w1a, o_w1aP, 64, 3, 32, 27);
    PW(w1b, o_w1bP, 64, 64, 64, 27);
    PW(w1d, o_w1dP, 64, 3, 32, 1);
    PW(wd1, o_wd1P, 128, 64, 64, 27);
    PW(w2a, o_w2aP, 128, 128, 128, 27);
    PW(w2b, o_w2bP, 128, 128, 128, 27);
    PW(wd2, o_wd2P, 256, 128, 128, 27);
    PW(w3a, o_w3aP, 256, 256, 256, 27);
    PW(w3b, o_w3bP, 256, 256, 256, 27);
    PW(wd3, o_wd3P, 128, 256, 256, 27);

    // ---- mask pyramids ----
    mask_down_kernel<<<(B * V2 + 255) / 256, 256, 0, stream>>>(m1, F(o_m2), B, 64);
    mask_down_kernel<<<(B * V3 + 255) / 256, 256, 0, stream>>>(F(o_m2), F(o_m3), B, 32);

    // ---- pack input ----
    pack_x_kernel<<<(B * V1 + 255) / 256, 256, 0, stream>>>(x, H(o_A0), B, V1);

    // grid: (voxel tiles / 8 waves, OC/32, batch); block = 256 (8 waves)
    // ---- stage 1 (64^3) ----
    conv3d_wmma_kernel<32, 64, 1, true, 0><<<dim3(V1 / 128, 2, B), 256, 0, stream>>>(
        H(o_A0), H(o_w1aP), m1, nullptr, nullptr, H(o_T1), 64);
    conv3d_wmma_kernel<64, 64, 1, true, 2><<<dim3(V1 / 128, 2, B), 256, 0, stream>>>(
        H(o_T1), H(o_w1bP), m1, H(o_A0), H(o_w1dP), H(o_X1), 64);
    conv3d_wmma_kernel<64, 128, 2, false, 0><<<dim3(V2 / 128, 4, B), 256, 0, stream>>>(
        H(o_X1), H(o_wd1P), nullptr, nullptr, nullptr, H(o_B2), 64);

    // ---- stage 2 (32^3) ----
    conv3d_wmma_kernel<128, 128, 1, true, 0><<<dim3(V2 / 128, 4, B), 256, 0, stream>>>(
        H(o_B2), H(o_w2aP), F(o_m2), nullptr, nullptr, H(o_T2), 32);
    conv3d_wmma_kernel<128, 128, 1, true, 1><<<dim3(V2 / 128, 4, B), 256, 0, stream>>>(
        H(o_T2), H(o_w2bP), F(o_m2), H(o_B2), nullptr, H(o_X2), 32);
    conv3d_wmma_kernel<128, 256, 2, false, 0><<<dim3(V3 / 128, 8, B), 256, 0, stream>>>(
        H(o_X2), H(o_wd2P), nullptr, nullptr, nullptr, H(o_B3), 32);

    // ---- stage 3 (16^3) ----
    conv3d_wmma_kernel<256, 256, 1, true, 0><<<dim3(V3 / 128, 8, B), 256, 0, stream>>>(
        H(o_B3), H(o_w3aP), F(o_m3), nullptr, nullptr, H(o_T3), 16);
    conv3d_wmma_kernel<256, 256, 1, true, 1><<<dim3(V3 / 128, 8, B), 256, 0, stream>>>(
        H(o_T3), H(o_w3bP), F(o_m3), H(o_B3), nullptr, H(o_X3), 16);
    conv3d_wmma_kernel<256, 128, 2, false, 0><<<dim3(V4 / 128, 4, B), 256, 0, stream>>>(
        H(o_X3), H(o_wd3P), nullptr, nullptr, nullptr, H(o_B4), 16);

    // ---- global mean+max pool -> [B, 256] f32 ----
    pool_kernel<<<dim3(128, B), 32, 0, stream>>>(H(o_B4), out, V4, 128);
}
